// MMPNType_57647051047693
// MI455X (gfx1250) — matrix operand order repaired and checked by reading, not compiled, not run on hardware
//
#include <hip/hip_runtime.h>
#include <hip/hip_bf16.h>

// ---------------------------------------------------------------------------
// Graph-network forward (4 chained LSTM scans) for MI455X / gfx1250, wave32.
//
// Key structure: the reference's LSTM scans run over edges(1280)/nodes(320)/
// graphs(64) with T=256 as an independent batch axis.  Each batch column is
// an independent recurrence, so 16 persistent workgroups each own 16 columns
// and run the whole scan in-kernel.  Per wave: one 16(col)x16(h) tile, cell
// state c held in WMMA f32 accumulators across all steps, h in LDS (f16),
// Whh resident in LDS (f16), Wih streamed from an f16/padded workspace copy
// (L2-resident).  segment_min is fused into the scan epilogue (relu >= 0 so
// plain fminf RMW into a +INF buffer; no atomics, no [E,T,H] materialization).
// ---------------------------------------------------------------------------

typedef __attribute__((ext_vector_type(16))) _Float16 v16h;
typedef __attribute__((ext_vector_type(8)))  _Float16 v8h;
typedef __attribute__((ext_vector_type(8)))  float    v8f;

#define TDIM   256
#define FNODE  64
#define FGLOB  16
#define HMSG   128
#define HUPD   128
#define HGRP   64
#define NNODES 320
#define NEDGES 1280
#define NGRAPH 64

__device__ __forceinline__ float sigf(float x){ return 1.0f/(1.0f + __expf(-x)); }

__device__ __forceinline__ v8f splat8(float x){
  v8f r;
#pragma unroll
  for (int i = 0; i < 8; ++i) r[i] = x;
  return r;
}

// A operand (16x32 f16): lane L (0..15) holds row M=L, K = base+[0..7] and
// base+[16..23]; lanes 16..31 hold the +8 K-offset half (caller passes aoff).
__device__ __forceinline__ v16h ldA(const _Float16* p){
  union { v16h v; v8h h[2]; } u;
  u.h[0] = *(const v8h*)(p);
  u.h[1] = *(const v8h*)(p + 16);
  return u.v;
}
// B operand (32x16 f16): lane L holds column N=L&15, K = base+[0..7] and
// base+[16..23] (lanes>=16 get K+8; caller folds that into p).
__device__ __forceinline__ v16h ldB(const _Float16* p){
  union { v16h v; v8h h[2]; } u;
  u.h[0] = *(const v8h*)(p);
  u.h[1] = *(const v8h*)(p + 16);
  return u.v;
}

__device__ __forceinline__ v8f wmma16(v16h a, v16h b, v8f c){
  return __builtin_amdgcn_wmma_f32_16x16x32_f16(false, a, false, b, (short)0, c,
                                                false, false);
}

// ---------------------------------------------------------------------------
// Generic persistent LSTM scan.  One WG = 16 batch columns.  H/16 waves, each
// wave owns one 16x16 (col x h) output tile; c stays in registers.
// z = x @ Wih^T + h @ Whh^T + (bih+bhh); torch gate order i,f,g,o.
// ---------------------------------------------------------------------------
template<int DIN_PAD, int H, class S, class E>
__device__ __forceinline__ void lstm_scan_core(int nsteps,
    const _Float16* __restrict__ wih,   // [4H][DIN_PAD] f16 (global, pre-padded)
    const float*    __restrict__ whhf,  // [4H][H] f32 (recurrent weights)
    const float*    __restrict__ bias,  // [4H] combined bih+bhh
    S stage, E epi,
    _Float16* whh_lds, _Float16* h_lds, _Float16* x_lds)
{
  static_assert(DIN_PAD % 32 == 0 && H % 32 == 0, "pad K to 32");
  const int tid  = threadIdx.x;
  const int nthr = blockDim.x;
  const int lane = tid & 31;
  const int wave = tid >> 5;
  const int h0   = wave * 16;          // this wave's h-tile base
  const int nsel = lane & 15;          // N index (h offset) / A row (column)
  const int hi   = lane >> 4;
  const int aoff = hi ? 8  : 0;        // A K-offset for upper lane half
  const int boff = hi ? 8 : 0;         // B K-offset for upper lane half

  // Park Whh in LDS as f16 (recurrent critical path), zero-init h.
  for (int i = tid; i < 4*H*H; i += nthr) whh_lds[i] = (_Float16)whhf[i];
  for (int i = tid; i < 16*H;  i += nthr) h_lds[i]   = (_Float16)0.0f;

  const float bg0 = bias[0*H + h0 + nsel];
  const float bg1 = bias[1*H + h0 + nsel];
  const float bg2 = bias[2*H + h0 + nsel];
  const float bg3 = bias[3*H + h0 + nsel];

  v8f c = splat8(0.0f);                // persistent cell state (16x16 tile)
  __syncthreads();

#pragma clang loop unroll(disable)
  for (int s = 0; s < nsteps; ++s){
    // ---- gather/concat this step's input into LDS (f32 -> f16, zero pad)
    for (int i = tid; i < 16*DIN_PAD; i += nthr){
      int col = i / DIN_PAD;
      int d   = i - col*DIN_PAD;
      x_lds[i] = (_Float16)stage(s, col, d);
    }
    if (s + 1 < nsteps) stage.prefetch(s + 1);
    __syncthreads();                   // x ready; prev step's h writes visible

    v8f a0 = splat8(bg0), a1 = splat8(bg1), a2 = splat8(bg2), a3 = splat8(bg3);

    // ---- input projection: K over DIN_PAD, Wih from global f16 (L2-hot)
    const _Float16* xrow  = x_lds + nsel * DIN_PAD;
    const _Float16* wbase = wih + (size_t)(h0 + nsel) * DIN_PAD + boff;
#pragma unroll
    for (int kc = 0; kc < DIN_PAD/32; ++kc){
      v16h a = ldA(xrow + kc*32 + aoff);
      a0 = wmma16(a, ldB(wbase + (size_t)0*H*DIN_PAD + kc*32), a0);
      a1 = wmma16(a, ldB(wbase + (size_t)1*H*DIN_PAD + kc*32), a1);
      a2 = wmma16(a, ldB(wbase + (size_t)2*H*DIN_PAD + kc*32), a2);
      a3 = wmma16(a, ldB(wbase + (size_t)3*H*DIN_PAD + kc*32), a3);
    }
    // ---- recurrent projection: K over H, Whh from LDS
    const _Float16* hrow = h_lds + nsel * H;
    const _Float16* wh   = whh_lds + (h0 + nsel) * H + boff;
#pragma unroll
    for (int kc = 0; kc < H/32; ++kc){
      v16h a = ldA(hrow + kc*32 + aoff);
      a0 = wmma16(a, ldB(wh + 0*H*H + kc*32), a0);
      a1 = wmma16(a, ldB(wh + 1*H*H + kc*32), a1);
      a2 = wmma16(a, ldB(wh + 2*H*H + kc*32), a2);
      a3 = wmma16(a, ldB(wh + 3*H*H + kc*32), a3);
    }
    // ---- cell update: gates share the D-tile layout, pure per-lane VALU
    float hreg[8];
#pragma unroll
    for (int v = 0; v < 8; ++v){
      float ig = sigf(a0[v]);
      float fg = sigf(a1[v]);
      float gg = tanhf(a2[v]);
      float og = sigf(a3[v]);
      float cv = fg * c[v] + ig * gg;
      c[v] = cv;
      hreg[v] = og * tanhf(cv);
    }
    __syncthreads();                   // all reads of old h/x complete
#pragma unroll
    for (int v = 0; v < 8; ++v){
      int col = v + 8*hi;              // D layout: M = v + 8*(lane>>4)
      h_lds[col*H + h0 + nsel] = (_Float16)hreg[v];
      epi(s, col, h0 + nsel, hreg[v]); // relu + store / fused segment-min
    }
  }
}

// ---------------------------- policies -------------------------------------
struct MsgStage {
  const float* nodes; const float* ga; const int* eidx; const int* bidx; int b0;
  __device__ float operator()(int e, int col, int d) const {
    int srcn = eidx[e];                              // faithful: src == tgt
    size_t b = (size_t)(b0 + col);
    if (d < 2*FNODE)
      return nodes[((size_t)srcn*TDIM + b)*FNODE + (d & (FNODE-1))];
    if (d < 2*FNODE + FGLOB)
      return ga[((size_t)bidx[srcn]*TDIM + b)*FGLOB + (d - 2*FNODE)];
    return 0.0f;
  }
  __device__ void prefetch(int e) const {
    int srcn = eidx[e];
    __builtin_prefetch(nodes + ((size_t)srcn*TDIM + b0)*FNODE, 0, 0);
  }
};
struct MsgEpi {
  float* aggr; const int* eidx; int b0;
  __device__ void operator()(int e, int col, int hh, float h) const {
    float m = fmaxf(h, 0.0f);
    size_t o = ((size_t)eidx[e]*TDIM + (size_t)(b0 + col))*HMSG + hh;
    aggr[o] = fminf(aggr[o], m);       // fused segment_min (WG owns columns)
  }
};

struct UpdStage {
  const float* nodes; const float* aggr; const float* ga; const int* bidx; int b0;
  __device__ float operator()(int n, int col, int d) const {
    size_t b = (size_t)(b0 + col);
    if (d < FNODE)        return nodes[((size_t)n*TDIM + b)*FNODE + d];
    if (d < FNODE + HMSG) return aggr[((size_t)n*TDIM + b)*HMSG + (d - FNODE)];
    if (d < FNODE + HMSG + FGLOB)
      return ga[((size_t)bidx[n]*TDIM + b)*FGLOB + (d - FNODE - HMSG)];
    return 0.0f;
  }
  __device__ void prefetch(int n) const {
    __builtin_prefetch(nodes + ((size_t)n*TDIM + b0)*FNODE, 0, 0);
  }
};
struct UpdEpi {
  float* upd; float* aggN; const int* bidx; int b0;
  __device__ void operator()(int n, int col, int hh, float h) const {
    float v = fmaxf(h, 0.0f);
    size_t b = (size_t)(b0 + col);
    upd[((size_t)n*TDIM + b)*HUPD + hh] = v;
    size_t o = ((size_t)bidx[n]*TDIM + b)*HUPD + hh;
    aggN[o] = fminf(aggN[o], v);
  }
};

struct GrpStage {
  const float* aggN; const float* ga; int b0;
  __device__ float operator()(int g, int col, int d) const {
    size_t b = (size_t)(b0 + col);
    if (d < HUPD)         return aggN[((size_t)g*TDIM + b)*HUPD + d];
    if (d < HUPD + FGLOB) return ga[((size_t)g*TDIM + b)*FGLOB + (d - HUPD)];
    return 0.0f;
  }
  __device__ void prefetch(int) const {}
};
struct GrpEpi {
  float* grp; int b0;
  __device__ void operator()(int g, int col, int hh, float h) const {
    grp[((size_t)g*TDIM + (size_t)(b0 + col))*HGRP + hh] = fmaxf(h, 0.0f);
  }
};

// ---------------------------- scan kernels ---------------------------------
__global__ void __launch_bounds__(256) msg_scan(
    const float* __restrict__ nodes, const float* __restrict__ ga,
    const int* __restrict__ eidx, const int* __restrict__ bidx,
    const _Float16* __restrict__ wih16, const float* __restrict__ whhf,
    const float* __restrict__ bias, float* __restrict__ aggr)
{
  extern __shared__ _Float16 smem[];
  _Float16* whh_lds = smem;                          // 4H*H
  _Float16* h_lds   = whh_lds + 4*HMSG*HMSG;         // 16*H
  _Float16* x_lds   = h_lds + 16*HMSG;               // 16*DIN_PAD
  MsgStage st{nodes, ga, eidx, bidx, (int)blockIdx.x*16};
  MsgEpi   ep{aggr, eidx, (int)blockIdx.x*16};
  lstm_scan_core<160, HMSG>(NEDGES, wih16, whhf, bias, st, ep,
                            whh_lds, h_lds, x_lds);
}

__global__ void __launch_bounds__(256) upd_scan(
    const float* __restrict__ nodes, const float* __restrict__ ga,
    const float* __restrict__ aggr, const int* __restrict__ bidx,
    const _Float16* __restrict__ wih16, const float* __restrict__ whhf,
    const float* __restrict__ bias,
    float* __restrict__ upd, float* __restrict__ aggN)
{
  extern __shared__ _Float16 smem[];
  _Float16* whh_lds = smem;
  _Float16* h_lds   = whh_lds + 4*HUPD*HUPD;
  _Float16* x_lds   = h_lds + 16*HUPD;
  UpdStage st{nodes, aggr, ga, bidx, (int)blockIdx.x*16};
  UpdEpi   ep{upd, aggN, bidx, (int)blockIdx.x*16};
  lstm_scan_core<224, HUPD>(NNODES, wih16, whhf, bias, st, ep,
                            whh_lds, h_lds, x_lds);
}

__global__ void __launch_bounds__(128) grp_scan(
    const float* __restrict__ aggN, const float* __restrict__ ga,
    const _Float16* __restrict__ wih16, const float* __restrict__ whhf,
    const float* __restrict__ bias, float* __restrict__ grp)
{
  extern __shared__ _Float16 smem[];
  _Float16* whh_lds = smem;
  _Float16* h_lds   = whh_lds + 4*HGRP*HGRP;
  _Float16* x_lds   = h_lds + 16*HGRP;
  GrpStage st{aggN, ga, (int)blockIdx.x*16};
  GrpEpi   ep{grp, (int)blockIdx.x*16};
  lstm_scan_core<160, HGRP>(NGRAPH, wih16, whhf, bias, st, ep,
                            whh_lds, h_lds, x_lds);
}

// ---- action head: only batch column T-1 survives ys[:, -1, :]; one wave ----
__global__ void act_head(const float* __restrict__ upd,
                         const float* __restrict__ aggN,
                         const float* __restrict__ grp,
                         const int* __restrict__ chosen_who,
                         const float* __restrict__ Wih, const float* __restrict__ Whh,
                         const float* __restrict__ bih, const float* __restrict__ bhh,
                         float* __restrict__ out)
{
  __shared__ float x[192];
  __shared__ float h[4], c[4], z[16];
  __shared__ float res[NGRAPH*4];
  const int t = threadIdx.x;
  if (t < 4){ h[t] = 0.0f; c[t] = 0.0f; }
  __syncthreads();
  for (int l = 0; l < NGRAPH; ++l){
    int cw = chosen_who[l];
    for (int d = t; d < 192; d += 32){
      float v;
      if (d < HUPD){
        if (cw == 3) v = aggN[((size_t)l*TDIM + (TDIM-1))*HUPD + d];
        else {
          int idx = (l == 0) ? cw : (cw + 5*l);    // adj==cw when cw!=3
          v = upd[((size_t)idx*TDIM + (TDIM-1))*HUPD + d];
        }
      } else v = grp[((size_t)l*TDIM + (TDIM-1))*HGRP + (d - HUPD)];
      x[d] = v;
    }
    __syncthreads();
    if (t < 16){
      float s = bih[t] + bhh[t];
      for (int d = 0; d < 192; ++d) s += Wih[t*192 + d]*x[d];
      for (int k = 0; k < 4; ++k)   s += Whh[t*4 + k]*h[k];
      z[t] = s;
    }
    __syncthreads();
    if (t < 4){
      float ig = sigf(z[t]), fg = sigf(z[4+t]);
      float gg = tanhf(z[8+t]), og = sigf(z[12+t]);
      c[t] = fg*c[t] + ig*gg;
      h[t] = og*tanhf(c[t]);
      res[l*4 + t] = h[t];
    }
    __syncthreads();
  }
  for (int l = t; l < NGRAPH; l += 32){
    float m = fmaxf(fmaxf(res[l*4], res[l*4+1]), fmaxf(res[l*4+2], res[l*4+3]));
    float e0 = __expf(res[l*4+0]-m), e1 = __expf(res[l*4+1]-m);
    float e2 = __expf(res[l*4+2]-m), e3 = __expf(res[l*4+3]-m);
    float s = e0+e1+e2+e3;
    out[l*4+0]=e0/s; out[l*4+1]=e1/s; out[l*4+2]=e2/s; out[l*4+3]=e3/s;
  }
}

// ---------------------------- prep kernels ---------------------------------
__global__ void convert_pad(const float* __restrict__ src, _Float16* __restrict__ dst,
                            int R, int C, int Cp)
{
  int i = blockIdx.x*blockDim.x + threadIdx.x;
  if (i >= R*Cp) return;
  int r = i / Cp, cc = i - r*Cp;
  dst[i] = (_Float16)((cc < C) ? src[r*C + cc] : 0.0f);
}
__global__ void combine_bias(const float* a, const float* b, float* d, int n){
  int i = blockIdx.x*blockDim.x + threadIdx.x;
  if (i < n) d[i] = a[i] + b[i];
}
__global__ void fill_val(float* p, float v, long n){
  long i = (long)blockIdx.x*blockDim.x + threadIdx.x;
  long st = (long)gridDim.x*blockDim.x;
  for (; i < n; i += st) p[i] = v;
}

// ---------------------------- launcher -------------------------------------
extern "C" void kernel_launch(void* const* d_in, const int* in_sizes, int n_in,
                              void* d_out, int out_size, void* d_ws, size_t ws_size,
                              hipStream_t stream)
{
  const float* nodes  = (const float*)d_in[0];
  const float* ga     = (const float*)d_in[1];
  const int*   eidx   = (const int*)d_in[2];      // row 0 == src == tgt (faithful)
  const int*   bidx   = (const int*)d_in[5];
  const int*   chosen = (const int*)d_in[6];
  const float *WihM=(const float*)d_in[7],  *WhhM=(const float*)d_in[8];
  const float *bihM=(const float*)d_in[9],  *bhhM=(const float*)d_in[10];
  const float *WihU=(const float*)d_in[11], *WhhU=(const float*)d_in[12];
  const float *bihU=(const float*)d_in[13], *bhhU=(const float*)d_in[14];
  const float *WihG=(const float*)d_in[15], *WhhG=(const float*)d_in[16];
  const float *bihG=(const float*)d_in[17], *bhhG=(const float*)d_in[18];
  const float *WihA=(const float*)d_in[19], *WhhA=(const float*)d_in[20];
  const float *bihA=(const float*)d_in[21], *bhhA=(const float*)d_in[22];

  char* w = (char*)d_ws;
  float* aggr = (float*)w; w += (size_t)NNODES*TDIM*HMSG*4;   // 42 MB
  float* aggN = (float*)w; w += (size_t)NGRAPH*TDIM*HUPD*4;   //  8 MB (adjacent to aggr)
  float* updb = (float*)w; w += (size_t)NNODES*TDIM*HUPD*4;   // 42 MB
  float* grpb = (float*)w; w += (size_t)NGRAPH*TDIM*HGRP*4;   //  4 MB
  _Float16* wihM16 = (_Float16*)w; w += (size_t)512*160*2;
  _Float16* wihU16 = (_Float16*)w; w += (size_t)512*224*2;
  _Float16* wihG16 = (_Float16*)w; w += (size_t)256*160*2;
  float* bm = (float*)w; w += 512*4;
  float* bu = (float*)w; w += 512*4;
  float* bg = (float*)w; w += 256*4;

  // aggr + aggN are adjacent: single +INF fill for both fused segment_mins.
  long nfill = (long)(NNODES + NGRAPH) * TDIM * HMSG;
  fill_val<<<1024, 256, 0, stream>>>(aggr, 3.4028235e38f, nfill);
  convert_pad<<<(512*160+255)/256, 256, 0, stream>>>(WihM, wihM16, 512, 144, 160);
  convert_pad<<<(512*224+255)/256, 256, 0, stream>>>(WihU, wihU16, 512, 208, 224);
  convert_pad<<<(256*160+255)/256, 256, 0, stream>>>(WihG, wihG16, 256, 144, 160);
  combine_bias<<<2, 256, 0, stream>>>(bihM, bhhM, bm, 512);
  combine_bias<<<2, 256, 0, stream>>>(bihU, bhhU, bu, 512);
  combine_bias<<<1, 256, 0, stream>>>(bihG, bhhG, bg, 256);

  size_t shM = (size_t)(4*HMSG*HMSG + 16*HMSG + 16*160) * 2;  // ~137 KB
  size_t shU = (size_t)(4*HUPD*HUPD + 16*HUPD + 16*224) * 2;  // ~139 KB
  size_t shG = (size_t)(4*HGRP*HGRP + 16*HGRP + 16*160) * 2;  // ~39 KB

  msg_scan<<<16, 256, shM, stream>>>(nodes, ga, eidx, bidx, wihM16, WhhM, bm, aggr);
  upd_scan<<<16, 256, shU, stream>>>(nodes, ga, aggr, bidx, wihU16, WhhU, bu, updb, aggN);
  grp_scan<<<16, 128, shG, stream>>>(aggN, ga, wihG16, WhhG, bg, grpb);
  act_head<<<1, 32, 0, stream>>>(updb, aggN, grpb, chosen, WihA, WhhA, bihA, bhhA,
                                 (float*)d_out);
  (void)in_sizes; (void)n_in; (void)out_size; (void)ws_size;
}
